// Hedgehog_20864951124431
// MI455X (gfx1250) — matrix operand order repaired and checked by reading, not compiled, not run on hardware
//
#include <hip/hip_runtime.h>

typedef __bf16 bf16;
typedef __attribute__((ext_vector_type(16))) __bf16 v16bf;
typedef __attribute__((ext_vector_type(8)))  __bf16 v8bf;
typedef __attribute__((ext_vector_type(8)))  float  v8f;

#define WMMA_BF16(a, b, c) \
  __builtin_amdgcn_wmma_f32_16x16x32_bf16(false, (a), false, (b), (short)0, (c), false, false)

static constexpr int Lh = 1024;   // sequence length
static constexpr int DM = 1024;   // model dim
static constexpr int NH = 16;     // heads
static constexpr int HD = 64;     // head dim
static constexpr int FD = 64;     // feature dim (per sign)
static constexpr int F2 = 128;    // 2*FD

// ---- fragment loaders (wave32 16-bit WMMA layouts, all K-contiguous) ----
// B fragment (32x16, K-major source): lane holds K = hlf*16 + 0..15 contiguous.
__device__ __forceinline__ v16bf ld_bfrag(const bf16* p) {
  return *(const v16bf*)p;
}
// A fragment (16x32, row K-contiguous): K = hlf*16 + 0..15 contiguous.
__device__ __forceinline__ v16bf ld_afrag(const bf16* row, int hlf) {
  v8bf lo = *(const v8bf*)(row + hlf * 16);
  v8bf hi = *(const v8bf*)(row + hlf * 16 + 8);
  return __builtin_shufflevector(lo, hi, 0, 1, 2, 3, 4, 5, 6, 7,
                                         8, 9, 10, 11, 12, 13, 14, 15);
}
// A fragment with only K=0..15 valid (padded to 32 with zeros).
__device__ __forceinline__ v16bf ld_afrag_k16(const bf16* row, int hlf) {
  v8bf lo = *(const v8bf*)(row);
  v8bf hi = *(const v8bf*)(row + 8);
  v16bf f;
  #pragma unroll
  for (int e = 0; e < 8; ++e) { f[e] = hlf ? (bf16)0.0f : lo[e]; f[8 + e] = hlf ? (bf16)0.0f : hi[e]; }
  return f;
}
// B fragment with only K=0..15 valid: hlf==1 lanes are all zero-pad.
__device__ __forceinline__ v16bf ld_bfrag_k16(const bf16* p, int hlf) {
  v16bf z;
  #pragma unroll
  for (int e = 0; e < 16; ++e) z[e] = (bf16)0.0f;
  return hlf ? z : *(const v16bf*)p;
}
// B fragment from 16 contiguous f32 (state in LDS), converted to bf16.
__device__ __forceinline__ v16bf ld_bfrag_f32(const float* p) {
  v16bf f;
  #pragma unroll
  for (int e = 0; e < 16; ++e) f[e] = (bf16)p[e];
  return f;
}

// ---- f32 -> bf16 conversion (plain) -------------------------------------
__global__ void cvt_f32_bf16(const float* __restrict__ src, bf16* __restrict__ dst, int n) {
  int i = blockIdx.x * blockDim.x + threadIdx.x;
  int stride = gridDim.x * blockDim.x;
  for (; i < n; i += stride) dst[i] = (bf16)src[i];
}

// ---- f32 -> bf16 conversion with per-matrix transpose -------------------
// src: H x R x C, dst: H x C x R
__global__ void cvt_t_f32_bf16(const float* __restrict__ src, bf16* __restrict__ dst,
                               int H, int R, int C) {
  int n = H * R * C;
  int i = blockIdx.x * blockDim.x + threadIdx.x;
  int stride = gridDim.x * blockDim.x;
  for (; i < n; i += stride) {
    int h = i / (R * C);
    int rem = i - h * (R * C);
    int r = rem / C, c = rem - r * C;
    dst[((size_t)h * C + c) * R + r] = (bf16)src[i];
  }
}

// ---- bf16 TN GEMM: C(MxN) = A(MxK) @ Bt(NxK)^T --------------------------
// Block tile 64x128, 4 waves; wave owns 32 N-cols -> 8 WMMA per 6 frag loads.
// OUTMODE: 0 = bf16 row-major, 1 = f32 row-major, 2 = bf16 transposed (NxM)
template <int OUTMODE>
__global__ __launch_bounds__(128) void gemm_bf16tn(const bf16* __restrict__ A,
                                                   const bf16* __restrict__ Bt,
                                                   void* __restrict__ Cv,
                                                   int M, int N, int K) {
  __shared__ alignas(64) bf16 sA[64 * 32];
  __shared__ alignas(64) bf16 sBt[128 * 32];
  const int tid  = threadIdx.x;
  const int wave = tid >> 5;
  const int lane = tid & 31;
  const int hlf  = lane >> 4;
  const int nn   = lane & 15;
  const int mm   = lane & 15;
  const int m0 = blockIdx.y * 64;
  const int n0 = blockIdx.x * 128;

  v8f acc[4][2] = {};

  const int r  = tid >> 1;           // staging rows
  const int cc = (tid & 1) * 16;     // 16-element (32 B) chunks

  for (int k0 = 0; k0 < K; k0 += 32) {
    *(v16bf*)&sA[r * 32 + cc]         = *(const v16bf*)&A[(size_t)(m0 + r) * K + k0 + cc];
    *(v16bf*)&sBt[r * 32 + cc]        = *(const v16bf*)&Bt[(size_t)(n0 + r) * K + k0 + cc];
    *(v16bf*)&sBt[(r + 64) * 32 + cc] = *(const v16bf*)&Bt[(size_t)(n0 + 64 + r) * K + k0 + cc];
    if (k0 + 32 < K) {  // warm L2/WGP$ for next k-tile (global_prefetch_b8)
      __builtin_prefetch(&A[(size_t)(m0 + r) * K + k0 + 32 + cc], 0, 3);
      __builtin_prefetch(&Bt[(size_t)(n0 + r) * K + k0 + 32 + cc], 0, 3);
      __builtin_prefetch(&Bt[(size_t)(n0 + 64 + r) * K + k0 + 32 + cc], 0, 3);
    }
    __syncthreads();

    v16bf bfrag0 = ld_bfrag(&sBt[(wave * 32 + nn) * 32 + hlf * 16]);
    v16bf bfrag1 = ld_bfrag(&sBt[(wave * 32 + 16 + nn) * 32 + hlf * 16]);
    #pragma unroll
    for (int mt = 0; mt < 4; ++mt) {
      v16bf afrag = ld_afrag(&sA[(mt * 16 + mm) * 32], hlf);
      acc[mt][0] = WMMA_BF16(afrag, bfrag0, acc[mt][0]);
      acc[mt][1] = WMMA_BF16(afrag, bfrag1, acc[mt][1]);
    }
    __syncthreads();
  }

  #pragma unroll
  for (int mt = 0; mt < 4; ++mt) {
    #pragma unroll
    for (int ct = 0; ct < 2; ++ct) {
      #pragma unroll
      for (int rr = 0; rr < 8; ++rr) {
        int row = m0 + mt * 16 + rr + 8 * hlf;
        int col = n0 + wave * 32 + ct * 16 + nn;
        if (OUTMODE == 0)      ((bf16*)Cv)[(size_t)row * N + col]  = (bf16)acc[mt][ct][rr];
        else if (OUTMODE == 1) ((float*)Cv)[(size_t)row * N + col] = acc[mt][ct][rr];
        else                   ((bf16*)Cv)[(size_t)col * M + row]  = (bf16)acc[mt][ct][rr];
      }
    }
  }
}

// ---- feature map: f = q_tile(16x64) @ Kmap[h](64x64); phi=[sm(f),sm(-f)] --
// KmapT is pre-transposed (H, FD, HD) so B-fragments are K(d)-contiguous.
__global__ __launch_bounds__(32) void featuremap_kernel(const bf16* __restrict__ QK,    // (L, DM)
                                                        const bf16* __restrict__ KmapT, // (H, FD, HD)
                                                        bf16* __restrict__ out,         // (H, L, F2)
                                                        bf16* __restrict__ outT) {      // (H, F2, L) or null
  __shared__ alignas(64) float sf[16 * 64];
  const int lane = threadIdx.x;
  const int hlf  = lane >> 4;
  const int nn   = lane & 15;
  const int mm   = lane & 15;
  const int h  = blockIdx.x >> 6;
  const int l0 = (blockIdx.x & 63) * 16;

  v8f acc[4] = {};
  #pragma unroll
  for (int s = 0; s < 2; ++s) {
    v16bf afrag = ld_afrag(&QK[(size_t)(l0 + mm) * DM + h * HD + s * 32], hlf);
    #pragma unroll
    for (int nt = 0; nt < 4; ++nt) {
      v16bf bfrag = ld_bfrag(&KmapT[(size_t)h * FD * HD + (size_t)(nt * 16 + nn) * HD + s * 32 + hlf * 16]);
      acc[nt] = WMMA_BF16(afrag, bfrag, acc[nt]);
    }
  }
  #pragma unroll
  for (int nt = 0; nt < 4; ++nt)
    #pragma unroll
    for (int rr = 0; rr < 8; ++rr)
      sf[(rr + 8 * hlf) * 64 + nt * 16 + nn] = acc[nt][rr];
  __syncthreads();

  const int row = lane & 15;
  const float sgn = hlf ? -1.0f : 1.0f;
  float mx = -1e30f;
  for (int j = 0; j < 64; ++j) mx = fmaxf(mx, sgn * sf[row * 64 + j]);
  float sum = 0.0f;
  for (int j = 0; j < 64; ++j) sum += __expf(sgn * sf[row * 64 + j] - mx);
  const float inv = 1.0f / sum;
  bf16* op = out + ((size_t)h * Lh + l0 + row) * F2 + hlf * FD;
  for (int j = 0; j < 64; ++j) {
    float p = __expf(sgn * sf[row * 64 + j] - mx) * inv;
    bf16 pv = (bf16)fmaxf(p, 1e-12f);
    op[j] = pv;
    if (outT) outT[((size_t)h * F2 + hlf * FD + j) * Lh + (l0 + row)] = pv;
  }
}

// ---- chunked causal linear attention, one block (4 waves) per head ------
__global__ __launch_bounds__(128) void hedgehog_attn(const bf16* __restrict__ Qf,  // (H, L, F2)
                                                     const bf16* __restrict__ Kf,  // (H, L, F2)
                                                     const bf16* __restrict__ KfT, // (H, F2, L)
                                                     const bf16* __restrict__ Vt,  // (DM, L): (h*64+d, l)
                                                     bf16* __restrict__ Y) {       // (L, DM)
  __shared__ alignas(64) float ST[HD * F2];   // state, (d, f) so f is contiguous
  __shared__ alignas(64) float ksumf[F2];
  __shared__ alignas(64) bf16 sQ[16 * F2];    // (t, f)
  __shared__ alignas(64) bf16 sK[16 * F2];    // (t, f)
  __shared__ alignas(64) bf16 sKt[F2 * 16];   // (f, t)
  __shared__ alignas(64) bf16 sVt[HD * 16];   // (d, t)
  __shared__ alignas(64) bf16 sA[4][16 * 16]; // per-wave masked attention tile
  __shared__ float zl[16];

  const int tid  = threadIdx.x;
  const int wave = tid >> 5;
  const int lane = tid & 31;
  const int hlf  = lane >> 4;
  const int nn   = lane & 15;
  const int mm   = lane & 15;
  const int h = blockIdx.x;

  for (int i = tid; i < HD * F2; i += 128) ST[i] = 0.0f;
  if (tid < F2) ksumf[tid] = 0.0f;
  __syncthreads();

  const bf16* qh = Qf + (size_t)h * Lh * F2;
  const bf16* kh = Kf + (size_t)h * Lh * F2;
  const bf16* kth = KfT + (size_t)h * F2 * Lh;

  for (int cch = 0; cch < Lh / 16; ++cch) {
    const int t0 = cch * 16;
    // wide cooperative staging
    *(v16bf*)&sQ[tid * 16]  = *(const v16bf*)&qh[(size_t)t0 * F2 + tid * 16];
    *(v16bf*)&sK[tid * 16]  = *(const v16bf*)&kh[(size_t)t0 * F2 + tid * 16];
    *(v16bf*)&sKt[tid * 16] = *(const v16bf*)&kth[(size_t)tid * Lh + t0];
    if (tid < HD)
      *(v16bf*)&sVt[tid * 16] = *(const v16bf*)&Vt[(size_t)(h * HD + tid) * Lh + t0];
    if (cch + 1 < Lh / 16) {  // prefetch next chunk (recurrence is latency-bound)
      __builtin_prefetch(&qh[(size_t)(t0 + 16) * F2 + tid * 16], 0, 3);
      __builtin_prefetch(&kh[(size_t)(t0 + 16) * F2 + tid * 16], 0, 3);
      __builtin_prefetch(&kth[(size_t)tid * Lh + t0 + 16], 0, 3);
    }
    __syncthreads();

    // A = Qf @ Kf^T (16x16), redundantly per wave; causal mask (keep s<=t)
    v8f accA = {};
    #pragma unroll
    for (int s = 0; s < 4; ++s) {
      v16bf afrag = ld_afrag(&sQ[mm * F2 + s * 32], hlf);
      v16bf bfrag = ld_bfrag(&sK[nn * F2 + s * 32 + hlf * 16]);
      accA = WMMA_BF16(afrag, bfrag, accA);
    }
    #pragma unroll
    for (int rr = 0; rr < 8; ++rr) {
      int m = rr + 8 * hlf;
      sA[wave][m * 16 + nn] = (bf16)((nn <= m) ? accA[rr] : 0.0f);
    }

    // z_t = qf_t . ksum_prev + rowsum(masked A)  (wave-0 lanes, same-wave LDS)
    if (tid < 16) {
      float z = 0.0f;
      for (int f = 0; f < F2; ++f) z += (float)sQ[tid * F2 + f] * ksumf[f];
      for (int s2 = 0; s2 < 16; ++s2) z += (float)sA[0][tid * 16 + s2];
      zl[tid] = z + 1e-12f;
    }

    // O = A_masked @ V + Qf @ S   (wave owns d-columns [16w, 16w+16))
    v8f accO = {};
    {
      v16bf afrag = ld_afrag_k16(&sA[wave][mm * 16], hlf);
      v16bf bfrag = ld_bfrag_k16(&sVt[(wave * 16 + nn) * 16], hlf);
      accO = WMMA_BF16(afrag, bfrag, accO);
    }
    #pragma unroll
    for (int s = 0; s < 4; ++s) {
      v16bf afrag = ld_afrag(&sQ[mm * F2 + s * 32], hlf);
      v16bf bfrag = ld_bfrag_f32(&ST[(wave * 16 + nn) * F2 + s * 32 + hlf * 16]);
      accO = WMMA_BF16(afrag, bfrag, accO);
    }
    __syncthreads();  // S reads done; zl visible to all waves

    #pragma unroll
    for (int rr = 0; rr < 8; ++rr) {
      int m = rr + 8 * hlf;
      Y[(size_t)(t0 + m) * DM + h * HD + wave * 16 + nn] = (bf16)(accO[rr] / zl[m]);
    }

    // state update: ST(d,f) += (V^T @ Kf)(d,f); wave owns f-cols [32w, 32w+32)
    #pragma unroll
    for (int dt = 0; dt < 4; ++dt) {
      v16bf afrag = ld_afrag_k16(&sVt[(dt * 16 + mm) * 16], hlf);
      #pragma unroll
      for (int ct = 0; ct < 2; ++ct) {
        const int f0 = wave * 32 + ct * 16;
        v16bf bfrag = ld_bfrag_k16(&sKt[(f0 + nn) * 16], hlf);
        v8f accD = {};
        accD = WMMA_BF16(afrag, bfrag, accD);
        #pragma unroll
        for (int rr = 0; rr < 8; ++rr)
          ST[(dt * 16 + rr + 8 * hlf) * F2 + f0 + nn] += accD[rr];
      }
    }
    if (tid < F2) {
      float s = ksumf[tid];
      for (int t = 0; t < 16; ++t) s += (float)sK[t * F2 + tid];
      ksumf[tid] = s;
    }
    __syncthreads();
  }
}

extern "C" void kernel_launch(void* const* d_in, const int* in_sizes, int n_in,
                              void* d_out, int out_size, void* d_ws, size_t ws_size,
                              hipStream_t stream) {
  (void)in_sizes; (void)n_in; (void)out_size; (void)ws_size;
  const float* x  = (const float*)d_in[0];
  const float* Wq = (const float*)d_in[1];
  const float* Wk = (const float*)d_in[2];
  const float* Wv = (const float*)d_in[3];
  const float* Wo = (const float*)d_in[4];
  const float* Kq = (const float*)d_in[5];
  const float* Kk = (const float*)d_in[6];

  char* ws = (char*)d_ws;
  size_t off = 0;
  auto carve = [&](size_t bytes) -> char* {
    char* p = ws + off;
    off += (bytes + 255) & ~(size_t)255;
    return p;
  };
  bf16* xb   = (bf16*)carve((size_t)Lh * DM * 2);
  bf16* wqt  = (bf16*)carve((size_t)DM * DM * 2);   // (n, k)
  bf16* wkt  = (bf16*)carve((size_t)DM * DM * 2);
  bf16* wvt  = (bf16*)carve((size_t)DM * DM * 2);
  bf16* wot  = (bf16*)carve((size_t)DM * DM * 2);
  bf16* kqT  = (bf16*)carve((size_t)NH * FD * HD * 2);  // (h, f, d)
  bf16* kkT  = (bf16*)carve((size_t)NH * FD * HD * 2);
  bf16* qb   = (bf16*)carve((size_t)Lh * DM * 2);   // (l, h*64+d)
  bf16* kb   = (bf16*)carve((size_t)Lh * DM * 2);
  bf16* vtb  = (bf16*)carve((size_t)DM * Lh * 2);   // (h*64+d, l)
  bf16* qfb  = (bf16*)carve((size_t)NH * Lh * F2 * 2);
  bf16* kfb  = (bf16*)carve((size_t)NH * Lh * F2 * 2);
  bf16* kftb = (bf16*)carve((size_t)NH * F2 * Lh * 2);
  bf16* yb   = (bf16*)carve((size_t)Lh * DM * 2);

  cvt_f32_bf16<<<512, 256, 0, stream>>>(x, xb, Lh * DM);
  cvt_t_f32_bf16<<<512, 256, 0, stream>>>(Wq, wqt, 1, DM, DM);
  cvt_t_f32_bf16<<<512, 256, 0, stream>>>(Wk, wkt, 1, DM, DM);
  cvt_t_f32_bf16<<<512, 256, 0, stream>>>(Wv, wvt, 1, DM, DM);
  cvt_t_f32_bf16<<<512, 256, 0, stream>>>(Wo, wot, 1, DM, DM);
  cvt_t_f32_bf16<<<64, 256, 0, stream>>>(Kq, kqT, NH, HD, FD);
  cvt_t_f32_bf16<<<64, 256, 0, stream>>>(Kk, kkT, NH, HD, FD);

  dim3 ggrid(DM / 128, Lh / 64);
  gemm_bf16tn<0><<<ggrid, 128, 0, stream>>>(xb, wqt, qb, Lh, DM, DM);
  gemm_bf16tn<0><<<ggrid, 128, 0, stream>>>(xb, wkt, kb, Lh, DM, DM);
  gemm_bf16tn<2><<<ggrid, 128, 0, stream>>>(xb, wvt, vtb, Lh, DM, DM);  // V transposed

  featuremap_kernel<<<NH * (Lh / 16), 32, 0, stream>>>(qb, kqT, qfb, nullptr);
  featuremap_kernel<<<NH * (Lh / 16), 32, 0, stream>>>(kb, kkT, kfb, kftb);

  hedgehog_attn<<<NH, 128, 0, stream>>>(qfb, kfb, kftb, vtb, yb);

  gemm_bf16tn<1><<<ggrid, 128, 0, stream>>>(yb, wot, d_out, Lh, DM, DM);
}
